// LinearAttention_91207925498415
// MI455X (gfx1250) — matrix operand order repaired and checked by reading, not compiled, not run on hardware
//
#include <hip/hip_runtime.h>

// ---------------------------------------------------------------------------
// Causal linear attention for MI455X (gfx1250), bf16 WMMA pipeline.
//   Zb  = bf16(Z)               (cvt_f32_to_bf16)
//   Qb  = Zb @ W_QK             (gemm_nt<0>)
//   Vb  = Zb @ W_V              (gemm_nt<0>)
//   Ab  = tril(Qb @ Zb^T) @ Vb  (attn_causal, flash-style streaming)
//   out = Ab @ W_O              (gemm_nt<1>)
// All matmuls via v_wmma_f32_16x16x32_bf16 with f32 accumulation.
// Tile staging uses GLOBAL_LOAD_ASYNC_TO_LDS_B128 (ASYNCcnt) when the
// toolchain exposes the builtin; plain copies otherwise.
// ---------------------------------------------------------------------------

typedef __bf16 bf16_t;
typedef __attribute__((ext_vector_type(16))) __bf16 v16bf;
typedef __attribute__((ext_vector_type(8)))  __bf16 v8bf;
typedef __attribute__((ext_vector_type(8)))  float  v8f;
typedef __attribute__((ext_vector_type(4)))  float  f32x4;
typedef __attribute__((ext_vector_type(4)))  int    v4i;

#define D_DIM 512
#define T_LEN 4096
#define B_BATCH 4
#define KSTEP 32
#define BLK_M 64
#define BLK_N 256
#define TQ 32
#define TK 32
#define ZLDK 520   // LDS row stride (bf16) for 512-wide tiles: 1040B, 16B-aligned
#define TLDK 40    // LDS row stride (bf16) for 32-wide GEMM tiles: 80B
#define SROW 36    // LDS row stride (f32) for 32-wide score tiles

// ---- async global->LDS copy (CDNA5 ASYNCcnt path), guarded ----------------
#if defined(__gfx1250__) && defined(__HIP_DEVICE_COMPILE__) && \
    __has_builtin(__builtin_amdgcn_global_load_async_to_lds_b128) && \
    __has_builtin(__builtin_amdgcn_s_wait_asynccnt)
#define USE_ASYNC 1
#else
#define USE_ASYNC 0
#endif

#if USE_ASYNC
typedef __attribute__((address_space(1))) v4i gas_v4i;  // global int4
typedef __attribute__((address_space(3))) v4i las_v4i;  // LDS int4
#endif

static __device__ inline void copy16_g2l(const bf16_t* g, bf16_t* l) {
#if USE_ASYNC
  __builtin_amdgcn_global_load_async_to_lds_b128((gas_v4i*)g, (las_v4i*)l, 0, 0);
#else
  *(v8bf*)l = *(const v8bf*)g;
#endif
}

static __device__ inline void async_wait_all() {
#if USE_ASYNC
  __builtin_amdgcn_s_wait_asynccnt(0);
#endif
}

// ---------------------------------------------------------------------------
static __device__ inline v16bf pack16(v8bf lo, v8bf hi) {
  return __builtin_shufflevector(lo, hi, 0,1,2,3,4,5,6,7,8,9,10,11,12,13,14,15);
}

static __device__ inline v8bf cvt8(f32x4 a, f32x4 b) {
  v8bf o;
#pragma unroll
  for (int j = 0; j < 4; ++j) { o[j] = (__bf16)a[j]; o[4 + j] = (__bf16)b[j]; }
  return o;
}

// A fragment (16x32 bf16, ISA 7.12.2) from a row-major bf16 tile, stride ldk.
// lane 0-15:  VGPR0-3 K=k0+0..7,  VGPR4-7 K=k0+16..23
// lane 16-31: VGPR0-3 K=k0+8..15, VGPR4-7 K=k0+24..31
static __device__ inline v16bf lds_loadA(const bf16_t* base, int row0, int k0,
                                         int ldk, int lane) {
  int m = lane & 15, hi = lane >> 4;
  const bf16_t* p = base + (size_t)(row0 + m) * ldk + k0 + 8 * hi;
  v8bf lo = *(const v8bf*)p;
  v8bf hh = *(const v8bf*)(p + 16);
  return pack16(lo, hh);
}

// B fragment (32x16 bf16) from a transposed [N][K] bf16 tile (row stride ldk):
// lane holds column N = n0+(lane&15), K = k0+8*(lane>=16)+0..7 and +16..23.
// Two 16B loads so padded (non-32B-multiple) strides stay legal.
static __device__ inline v16bf lds_loadBT(const bf16_t* baseT, int n0, int k0,
                                          int ldk, int lane) {
  int nn = lane & 15, hi = lane >> 4;
  const bf16_t* p = baseT + (size_t)(n0 + nn) * ldk + k0 + 8 * hi;
  v8bf lo = *(const v8bf*)p;
  v8bf hh = *(const v8bf*)(p + 16);
  return pack16(lo, hh);
}

// B fragment from a row-major [K][N] bf16 tile (row stride ldn): strided gather.
static __device__ inline v16bf lds_loadB_gather(const bf16_t* base, int k0,
                                                int n0, int ldn, int lane) {
  int nn = lane & 15, hi = lane >> 4;
  const bf16_t* p = base + (size_t)(k0 + 8 * hi) * ldn + n0 + nn;
  v16bf r;
#pragma unroll
  for (int i = 0; i < 16; ++i) r[i] = p[(size_t)(i + (i >= 8 ? 8 : 0)) * ldn];
  return r;
}

static __device__ inline v8f wmma_bf16(v16bf a, v16bf b, v8f c) {
  return __builtin_amdgcn_wmma_f32_16x16x32_bf16(false, a, false, b,
                                                 (short)0, c, false, false);
}

// ---------------------------------------------------------------------------
// Phase 0: elementwise f32 -> bf16 (8 elements / thread)
// ---------------------------------------------------------------------------
__global__ __launch_bounds__(256)
void cvt_f32_to_bf16(const float* __restrict__ src, bf16_t* __restrict__ dst) {
  const size_t i = ((size_t)blockIdx.x * blockDim.x + threadIdx.x) * 8;
  f32x4 a = *(const f32x4*)(src + i);
  f32x4 b = *(const f32x4*)(src + i + 4);
  *(v8bf*)(dst + i) = cvt8(a, b);
}

// ---------------------------------------------------------------------------
// GEMM: C[M x 512] = A_bf16[M x 512] @ B_f32[512 x 512]
// Workgroup: 256 threads (8 waves), tile 64x256, K-step 32.
// Wave w: rows [32*(w>>2), +32), cols [64*(w&3), +64). 8 wmma per K-step.
// ---------------------------------------------------------------------------
template <int OUTF32>
__global__ __launch_bounds__(256)
void gemm_nt(const bf16_t* __restrict__ Ap, const float* __restrict__ Bp,
             void* __restrict__ Cp) {
  __shared__ bf16_t At[BLK_M * TLDK];   // [64][40] row-major
  __shared__ bf16_t Bt[BLK_N * TLDK];   // [256][40] transposed: Bt[n][k]

  const int tid = threadIdx.x;
  const int lane = tid & 31, w = tid >> 5;
  const int wm = w >> 2, wn = w & 3;
  const int m0 = blockIdx.x * BLK_M;
  const int n0 = blockIdx.y * BLK_N;

  v8f acc[2][4];
#pragma unroll
  for (int mt = 0; mt < 2; ++mt)
#pragma unroll
    for (int nt = 0; nt < 4; ++nt) acc[mt][nt] = (v8f)0.0f;

  for (int kk = 0; kk < D_DIM; kk += KSTEP) {
    __syncthreads();
    // --- stage A tile (async global->LDS, 16B per thread) ---
    {
      const int r = tid >> 2;
      const int c = (tid & 3) * 8;
      copy16_g2l(Ap + (size_t)(m0 + r) * D_DIM + kk + c, At + r * TLDK + c);
    }
    // --- stage B tile transposed (f32 -> bf16 conversion) ---
    {
      const int k = tid >> 3;
      const int c0 = (tid & 7) * 32;
      const float* s = Bp + (size_t)(kk + k) * D_DIM + n0 + c0;
#pragma unroll
      for (int i = 0; i < 8; ++i) {
        f32x4 x = *(const f32x4*)(s + 4 * i);
#pragma unroll
        for (int j = 0; j < 4; ++j)
          Bt[(size_t)(c0 + 4 * i + j) * TLDK + k] = (__bf16)x[j];
      }
    }
    async_wait_all();
    __syncthreads();

    v16bf afrag[2];
#pragma unroll
    for (int mt = 0; mt < 2; ++mt)
      afrag[mt] = lds_loadA(At, 32 * wm + 16 * mt, 0, TLDK, lane);
#pragma unroll
    for (int nt = 0; nt < 4; ++nt) {
      v16bf bfrag = lds_loadBT(Bt, 64 * wn + 16 * nt, 0, TLDK, lane);
#pragma unroll
      for (int mt = 0; mt < 2; ++mt)
        acc[mt][nt] = wmma_bf16(afrag[mt], bfrag, acc[mt][nt]);
    }
  }

  // --- epilogue: C layout = VGPR r -> M = r + 8*(lane>=16), N = lane&15 ---
  const int nn = lane & 15, hi = lane >> 4;
#pragma unroll
  for (int mt = 0; mt < 2; ++mt)
#pragma unroll
    for (int nt = 0; nt < 4; ++nt)
#pragma unroll
      for (int r = 0; r < 8; ++r) {
        const size_t row = (size_t)m0 + 32 * wm + 16 * mt + r + 8 * hi;
        const size_t col = (size_t)n0 + 64 * wn + 16 * nt + nn;
        if (OUTF32)
          ((float*)Cp)[row * D_DIM + col] = acc[mt][nt][r];
        else
          ((bf16_t*)Cp)[row * D_DIM + col] = (__bf16)acc[mt][nt][r];
      }
}

// ---------------------------------------------------------------------------
// attn_causal: one workgroup = (batch b, 32 query rows). 8 waves.
// For each key block kb <= qb:
//   - async-stage Zk[32][512] and Vk[32][512] (bf16, row-major, padded) in LDS
//   - wave w computes partial S over its D-slice [64w,64w+64) from
//     register-held Q fragments; partials deterministically summed in fixed order
//     in LDS (no FP atomics -> bit-stable across graph replays)
//   - causal mask on diagonal block applied while reloading S as bf16
//     A fragments; then S(32x32) @ V(32x64 per wave) accumulates f32.
// Result written as bf16 to Ab (workspace).
// ---------------------------------------------------------------------------
__global__ __launch_bounds__(256)
void attn_causal(const bf16_t* __restrict__ Zb, const bf16_t* __restrict__ Qb,
                 const bf16_t* __restrict__ Vb, bf16_t* __restrict__ Ab) {
  __shared__ bf16_t Zk[TK * ZLDK];        // keys,   [s][d] row-major
  __shared__ bf16_t Vk[TK * ZLDK];        // values, [s][col] row-major
  __shared__ float  Sp[8 * TQ * SROW];    // per-wave partial S

  const int tid = threadIdx.x;
  const int lane = tid & 31, w = tid >> 5;
  const int nn = lane & 15, hi = lane >> 4;
  const int qb = blockIdx.x;              // query block in [0,128)
  const int b  = blockIdx.y;              // batch
  const int t0 = qb * TQ;
  const size_t rowQ0 = (size_t)b * T_LEN + t0;

  // Q fragments for this wave's D-slice [64w, 64w+64): [mtile][kstep]
  v16bf qf[2][2];
#pragma unroll
  for (int mt = 0; mt < 2; ++mt)
#pragma unroll
    for (int kk = 0; kk < 2; ++kk) {
      const bf16_t* p = Qb + (rowQ0 + 16 * mt + nn) * D_DIM +
                        64 * w + 32 * kk + 8 * hi;
      v8bf lo = *(const v8bf*)p;
      v8bf hh = *(const v8bf*)(p + 16);
      qf[mt][kk] = pack16(lo, hh);
    }

  v8f oacc[2][4];
#pragma unroll
  for (int mt = 0; mt < 2; ++mt)
#pragma unroll
    for (int nt = 0; nt < 4; ++nt) oacc[mt][nt] = (v8f)0.0f;

  for (int kb = 0; kb <= qb; ++kb) {
    __syncthreads();  // prior-iteration readers done before overwriting LDS
    const size_t rowK0 = (size_t)b * T_LEN + kb * TK;

    // --- async-stage Zk and Vk (128B per thread per tile, contiguous) ---
    {
      const int r = tid >> 3;
      const int c0 = (tid & 7) * 64;
      const bf16_t* zs = Zb + (rowK0 + r) * D_DIM + c0;
      const bf16_t* vs = Vb + (rowK0 + r) * D_DIM + c0;
      bf16_t* zd = Zk + r * ZLDK + c0;
      bf16_t* vd = Vk + r * ZLDK + c0;
#pragma unroll
      for (int i = 0; i < 8; ++i) {
        copy16_g2l(zs + 8 * i, zd + 8 * i);
        copy16_g2l(vs + 8 * i, vd + 8 * i);
      }
    }
    async_wait_all();
    __syncthreads();

    // --- partial scores over this wave's 64-wide D slice ---
    v8f sacc[2][2];
#pragma unroll
    for (int mt = 0; mt < 2; ++mt)
#pragma unroll
      for (int nt = 0; nt < 2; ++nt) sacc[mt][nt] = (v8f)0.0f;
#pragma unroll
    for (int kk = 0; kk < 2; ++kk) {
      v16bf bfr[2];
#pragma unroll
      for (int nt = 0; nt < 2; ++nt)
        bfr[nt] = lds_loadBT(Zk, 16 * nt, 64 * w + 32 * kk, ZLDK, lane);
#pragma unroll
      for (int mt = 0; mt < 2; ++mt)
#pragma unroll
        for (int nt = 0; nt < 2; ++nt)
          sacc[mt][nt] = wmma_bf16(qf[mt][kk], bfr[nt], sacc[mt][nt]);
    }
    // write partials (C layout -> row-major [32][SROW] region per wave)
    {
      float* myS = Sp + w * (TQ * SROW);
#pragma unroll
      for (int mt = 0; mt < 2; ++mt)
#pragma unroll
        for (int nt = 0; nt < 2; ++nt)
#pragma unroll
          for (int r = 0; r < 8; ++r)
            myS[(16 * mt + r + 8 * hi) * SROW + 16 * nt + nn] = sacc[mt][nt][r];
    }
    __syncthreads();
    // --- deterministic reduction of 8 partials into Sp region 0 ---
    {
      const int e0 = tid * 4;              // 1024 logical elements
      const int row = e0 >> 5;
      const int col = e0 & 31;
      const int off = row * SROW + col;
      float s0 = 0.f, s1 = 0.f, s2 = 0.f, s3 = 0.f;
#pragma unroll
      for (int ww = 0; ww < 8; ++ww) {
        const float* p = Sp + ww * (TQ * SROW) + off;
        s0 += p[0]; s1 += p[1]; s2 += p[2]; s3 += p[3];
      }
      Sp[off] = s0; Sp[off + 1] = s1; Sp[off + 2] = s2; Sp[off + 3] = s3;
    }
    __syncthreads();

    // --- masked S -> bf16 A fragments, then oacc += S @ V ---
    const bool diag = (kb == qb);
    v16bf sfr[2];
#pragma unroll
    for (int mt = 0; mt < 2; ++mt) {
      const int row = 16 * mt + nn;
      const int q = t0 + row;
      const float* p = Sp + row * SROW;
      v16bf r;
#pragma unroll
      for (int i = 0; i < 8; ++i) {
        const int k1 = 8 * hi + i;        // K = k1  (VGPR 0-3 group)
        const int k2 = 8 * hi + 16 + i;   // K = k2  (VGPR 4-7 group)
        float x1 = p[k1];
        float x2 = p[k2];
        if (diag) {
          if (kb * TK + k1 > q) x1 = 0.f;
          if (kb * TK + k2 > q) x2 = 0.f;
        }
        r[i] = (__bf16)x1;
        r[8 + i] = (__bf16)x2;
      }
      sfr[mt] = r;
    }
#pragma unroll
    for (int nt = 0; nt < 4; ++nt) {
      v16bf vfr = lds_loadB_gather(Vk, 0, 64 * w + 16 * nt, ZLDK, lane);
#pragma unroll
      for (int mt = 0; mt < 2; ++mt)
        oacc[mt][nt] = wmma_bf16(sfr[mt], vfr, oacc[mt][nt]);
    }
  }

  // --- store attn tile as bf16 ---
#pragma unroll
  for (int mt = 0; mt < 2; ++mt)
#pragma unroll
    for (int nt = 0; nt < 4; ++nt)
#pragma unroll
      for (int r = 0; r < 8; ++r) {
        const size_t row = rowQ0 + 16 * mt + r + 8 * hi;
        const int col = 64 * w + 16 * nt + nn;
        Ab[row * D_DIM + col] = (__bf16)oacc[mt][nt][r];
      }
}

// ---------------------------------------------------------------------------
extern "C" void kernel_launch(void* const* d_in, const int* in_sizes, int n_in,
                              void* d_out, int out_size, void* d_ws,
                              size_t ws_size, hipStream_t stream) {
  (void)in_sizes; (void)n_in; (void)out_size; (void)ws_size;
  const float* Z   = (const float*)d_in[0];  // (4,4096,512)
  const float* Wqk = (const float*)d_in[1];  // (512,512)
  const float* Wv  = (const float*)d_in[2];  // (512,512)
  const float* Wo  = (const float*)d_in[3];  // (512,512)
  float* out = (float*)d_out;                // (4,4096,512) f32

  const size_t M = (size_t)B_BATCH * T_LEN;            // 16384
  const size_t BUF_BYTES = M * D_DIM * sizeof(bf16_t); // 16 MB each
  char* ws = (char*)d_ws;
  bf16_t* Zb = (bf16_t*)(ws);
  bf16_t* Qb = (bf16_t*)(ws + BUF_BYTES);
  bf16_t* Vb = (bf16_t*)(ws + 2 * BUF_BYTES);
  bf16_t* Ab = (bf16_t*)(ws + 3 * BUF_BYTES);

  dim3 blk(256);

  // Phase 0: Z -> bf16
  const unsigned cvtBlocks = (unsigned)((M * D_DIM) / (256 * 8));  // 4096
  cvt_f32_to_bf16<<<cvtBlocks, blk, 0, stream>>>(Z, Zb);

  // Phase 1: Qb, Vb
  dim3 gemmGrid((unsigned)(M / BLK_M), D_DIM / BLK_N);  // (256, 2)
  gemm_nt<0><<<gemmGrid, blk, 0, stream>>>(Zb, Wqk, (void*)Qb);
  gemm_nt<0><<<gemmGrid, blk, 0, stream>>>(Zb, Wv,  (void*)Vb);

  // Phase 2: streaming causal attention
  dim3 attnGrid(T_LEN / TQ, B_BATCH);                   // (128, 4)
  attn_causal<<<attnGrid, blk, 0, stream>>>(Zb, Qb, Vb, Ab);

  // Phase 3: out = Ab @ W_O
  gemm_nt<1><<<gemmGrid, blk, 0, stream>>>(Ab, Wo, (void*)out);
}
